// MSRSA_44890998178157
// MI455X (gfx1250) — compile-verified, matrix operand order checked by reading, not run on hardware
//
#include <hip/hip_runtime.h>

// MSRSA attention for MI455X (gfx1250, wave32, WMMA).
// B=2 H=16 L=2048 D=128. Outputs: out (B,H,L,D) then attn_weight (B,H,L,L), f32.
// Two-pass online softmax per 16-query wave tile; bf16 WMMA for QK^T and PV;
// register-double-buffered global->LDS staging; explicit 4-way WMMA interleave.

#define B_ 2
#define H_ 16
#define L_ 2048
#define D_ 128
#define KT 64
#define NWAVE 8

typedef __attribute__((ext_vector_type(16))) __bf16 v16bf;
typedef __attribute__((ext_vector_type(8)))  __bf16 v8bf;
typedef __attribute__((ext_vector_type(8)))  float  v8f;
typedef __attribute__((ext_vector_type(4)))  unsigned short us4;

static __device__ inline unsigned short f2bfu(float f) {
  unsigned u = __builtin_bit_cast(unsigned, f);
  unsigned r = u + 0x7FFFu + ((u >> 16) & 1u);   // round-to-nearest-even
  return (unsigned short)(r >> 16);
}
static __device__ inline __bf16 u2bf(unsigned short u) {
  return __builtin_bit_cast(__bf16, u);
}
static __device__ inline v16bf cat16(v8bf lo, v8bf hi) {
  return __builtin_shufflevector(lo, hi, 0,1,2,3,4,5,6,7,8,9,10,11,12,13,14,15);
}
static __device__ inline v8f wmma_bf16(v16bf a, v16bf b, v8f c) {
  // D = A(16x32 bf16) * B(32x16 bf16) + C(16x16 f32)
  return __builtin_amdgcn_wmma_f32_16x16x32_bf16(false, a, false, b, (short)0, c,
                                                 false, false);
}

// Fetch a 64x128 f32 tile (rows kt..kt+63) into 8 float4 registers per thread.
static __device__ inline void load_tile(const float* __restrict__ base, int kt,
                                        int tid, float4 f[8]) {
#pragma unroll
  for (int i = 0; i < 8; ++i) {
    int u  = tid + 256 * i;        // 2048 float4 units, coalesced
    int kr = u >> 5;               // 0..63
    int c4 = (u & 31) * 4;         // 0..124
    f[i] = *(const float4*)(base + (size_t)(kt + kr) * D_ + c4);
  }
}
// Convert + store a staged K tile into LDS (row-major bf16).
static __device__ inline void store_K(unsigned short (*sK)[D_ + 8], int tid,
                                      const float4 f[8]) {
#pragma unroll
  for (int i = 0; i < 8; ++i) {
    int u  = tid + 256 * i;
    int kr = u >> 5;
    int c4 = (u & 31) * 4;
    us4 pk;
    pk[0] = f2bfu(f[i].x); pk[1] = f2bfu(f[i].y);
    pk[2] = f2bfu(f[i].z); pk[3] = f2bfu(f[i].w);
    *(us4*)&sK[kr][c4] = pk;
  }
}
// Convert + store a staged V tile into LDS transposed (d-major bf16).
static __device__ inline void store_Vt(unsigned short (*sVt)[KT + 8], int tid,
                                       const float4 f[8]) {
#pragma unroll
  for (int i = 0; i < 8; ++i) {
    int u  = tid + 256 * i;
    int kr = u >> 5;
    int c4 = (u & 31) * 4;
    sVt[c4 + 0][kr] = f2bfu(f[i].x);
    sVt[c4 + 1][kr] = f2bfu(f[i].y);
    sVt[c4 + 2][kr] = f2bfu(f[i].z);
    sVt[c4 + 3][kr] = f2bfu(f[i].w);
  }
}

__global__ __launch_bounds__(256) void msrsa_attn(
    const float* __restrict__ Q, const float* __restrict__ K,
    const float* __restrict__ V, const float* __restrict__ Adj,
    const float* __restrict__ Ds, const float* __restrict__ wA,
    const float* __restrict__ wD, float* __restrict__ out)
{
  // LDS staging (padded strides keep 16B alignment, break bank conflicts)
  __shared__ __align__(16) unsigned short sK[KT][D_ + 8];     // 64 x 136 bf16
  __shared__ __align__(16) unsigned short sVt[D_][KT + 8];    // 128 x 72 bf16
  __shared__ __align__(16) unsigned short sP[NWAVE][16][40];  // per-wave P buffer

  const int tid  = threadIdx.x;
  const int wid  = tid >> 5;
  const int lane = tid & 31;
  const int n    = lane & 15;   // A-operand row / B-operand column / C column
  const int hh   = lane >> 4;   // half-wave selects K-dim halves / C row-halves

  const int b = blockIdx.z;
  const int h = blockIdx.y;
  const int qbase = blockIdx.x * (NWAVE * 16) + wid * 16;

  const float wa = wA[h];
  const float wd = wD[h];
  const float inv_scale = 0.08838834764831843f;  // 1/sqrt(128)

  const size_t qkv_base = ((size_t)(b * H_ + h)) * L_ * D_;
  const size_t ad_base  = ((size_t)b) * L_ * L_;
  const float* Kb = K + qkv_base;
  const float* Vb = V + qkv_base;

  // ---- Q tile (16x128) -> bf16 A-operand registers (4 chunks of 16x32) ----
  v16bf qa[4];
  {
    const float* qrow = Q + qkv_base + (size_t)(qbase + n) * D_;
#pragma unroll
    for (int c = 0; c < 4; ++c) {
      const float* p0 = qrow + 32 * c + 8 * hh;        // K = 8h .. 8h+7
      const float* p1 = qrow + 32 * c + 16 + 8 * hh;   // K = 16+8h .. 16+8h+7
      v16bf a;
#pragma unroll
      for (int j = 0; j < 8; ++j) a[j] = u2bf(f2bfu(p0[j]));
#pragma unroll
      for (int j = 0; j < 8; ++j) a[8 + j] = u2bf(f2bfu(p1[j]));
      qa[c] = a;
    }
  }

  float mrow[8], lrow[8];
#pragma unroll
  for (int r = 0; r < 8; ++r) { mrow[r] = -__builtin_huge_valf(); lrow[r] = 0.0f; }

  float4 kbuf[8], vbuf[8];

  // ================= PASS 1: row max + sum(exp) =================
  load_tile(Kb, 0, tid, kbuf);                        // prefetch tile 0
  for (int kt = 0; kt < L_; kt += KT) {
    __syncthreads();                                  // prev compute done with sK
    store_K(sK, tid, kbuf);
    if (kt + KT < L_) load_tile(Kb, kt + KT, tid, kbuf);  // overlap with compute
    __syncthreads();                                  // sK tile ready

    // prefetch modulation factors for the whole KT chunk (64 loads in flight,
    // consumed only after the WMMA block)
    float fmod[4][8];
#pragma unroll
    for (int ks = 0; ks < 4; ++ks) {
      const int kg = kt + ks * 16 + n;
#pragma unroll
      for (int r = 0; r < 8; ++r) {
        size_t off = ad_base + (size_t)(qbase + hh * 8 + r) * L_ + kg;
        fmod[ks][r] = 1.0f + Adj[off] * wa + Ds[off] * wd;
      }
    }

    // 4 independent S-tile accumulator chains, c-outer for WMMA ILP
    v8f s[4];
#pragma unroll
    for (int ks = 0; ks < 4; ++ks) s[ks] = v8f{};
#pragma unroll
    for (int c = 0; c < 4; ++c) {
      v16bf kb[4];
#pragma unroll
      for (int ks = 0; ks < 4; ++ks) {
        v8bf lo = *(const v8bf*)&sK[ks * 16 + n][32 * c + 8 * hh];
        v8bf hi = *(const v8bf*)&sK[ks * 16 + n][32 * c + 16 + 8 * hh];
        kb[ks] = cat16(lo, hi);
      }
#pragma unroll
      for (int ks = 0; ks < 4; ++ks) s[ks] = wmma_bf16(qa[c], kb[ks], s[ks]);
    }

#pragma unroll
    for (int ks = 0; ks < 4; ++ks) {
      float sv[8];
#pragma unroll
      for (int r = 0; r < 8; ++r) sv[r] = s[ks][r] * fmod[ks][r] * inv_scale;
#pragma unroll
      for (int r = 0; r < 8; ++r) {
        float tm = sv[r];
        tm = fmaxf(tm, __shfl_xor(tm, 1, 32));
        tm = fmaxf(tm, __shfl_xor(tm, 2, 32));
        tm = fmaxf(tm, __shfl_xor(tm, 4, 32));
        tm = fmaxf(tm, __shfl_xor(tm, 8, 32));
        float mn = fmaxf(mrow[r], tm);
        float e  = __expf(sv[r] - mn);
        float ts = e;
        ts += __shfl_xor(ts, 1, 32);
        ts += __shfl_xor(ts, 2, 32);
        ts += __shfl_xor(ts, 4, 32);
        ts += __shfl_xor(ts, 8, 32);
        lrow[r] = lrow[r] * __expf(mrow[r] - mn) + ts;
        mrow[r] = mn;
      }
    }
  }

  float linv[8];
#pragma unroll
  for (int r = 0; r < 8; ++r) linv[r] = 1.0f / lrow[r];

  // ================= PASS 2: P write + O = P @ V =================
  v8f acc[8];
#pragma unroll
  for (int t = 0; t < 8; ++t) acc[t] = v8f{};

  float* outO = out + qkv_base;
  float* outW = out + (size_t)B_ * H_ * L_ * D_ + ((size_t)(b * H_ + h)) * L_ * L_;

  load_tile(Kb, 0, tid, kbuf);                        // prefetch tile 0
  load_tile(Vb, 0, tid, vbuf);
  for (int kt = 0; kt < L_; kt += KT) {
    __syncthreads();
    store_K(sK, tid, kbuf);
    store_Vt(sVt, tid, vbuf);
    if (kt + KT < L_) {
      load_tile(Kb, kt + KT, tid, kbuf);              // overlap with compute
      load_tile(Vb, kt + KT, tid, vbuf);
    }
    __syncthreads();

    // prefetch modulation factors
    float fmod[4][8];
#pragma unroll
    for (int ks = 0; ks < 4; ++ks) {
      const int kg = kt + ks * 16 + n;
#pragma unroll
      for (int r = 0; r < 8; ++r) {
        size_t off = ad_base + (size_t)(qbase + hh * 8 + r) * L_ + kg;
        fmod[ks][r] = 1.0f + Adj[off] * wa + Ds[off] * wd;
      }
    }

    // all 4 S tiles, interleaved WMMA chains
    v8f s[4];
#pragma unroll
    for (int ks = 0; ks < 4; ++ks) s[ks] = v8f{};
#pragma unroll
    for (int c = 0; c < 4; ++c) {
      v16bf kb[4];
#pragma unroll
      for (int ks = 0; ks < 4; ++ks) {
        v8bf lo = *(const v8bf*)&sK[ks * 16 + n][32 * c + 8 * hh];
        v8bf hi = *(const v8bf*)&sK[ks * 16 + n][32 * c + 16 + 8 * hh];
        kb[ks] = cat16(lo, hi);
      }
#pragma unroll
      for (int ks = 0; ks < 4; ++ks) s[ks] = wmma_bf16(qa[c], kb[ks], s[ks]);
    }

#pragma unroll
    for (int kk = 0; kk < 2; ++kk) {        // 32-key chunk for PV contraction
#pragma unroll
      for (int ks = 0; ks < 2; ++ks) {      // two 16-key S tiles per chunk
        const int ksub = kk * 2 + ks;
        const int kg = kt + ksub * 16 + n;
#pragma unroll
        for (int r = 0; r < 8; ++r) {
          size_t off = (size_t)(qbase + hh * 8 + r) * L_ + kg;
          float sc = s[ksub][r] * fmod[ksub][r] * inv_scale;
          float p  = __expf(sc - mrow[r]) * linv[r];
          outW[off] = p;                                        // attn_weight
          sP[wid][hh * 8 + r][ks * 16 + n] = f2bfu(p);          // stage for PV
        }
      }
      // P 16x32 A-operand (same-wave LDS RAW; hw DScnt keeps order)
      v8bf plo = *(const v8bf*)&sP[wid][n][8 * hh];
      v8bf phi = *(const v8bf*)&sP[wid][n][16 + 8 * hh];
      v16bf pa = cat16(plo, phi);
#pragma unroll
      for (int t = 0; t < 8; ++t) {
        v8bf vlo = *(const v8bf*)&sVt[16 * t + n][kk * 32 + 8 * hh];
        v8bf vhi = *(const v8bf*)&sVt[16 * t + n][kk * 32 + 16 + 8 * hh];
        acc[t] = wmma_bf16(pa, cat16(vlo, vhi), acc[t]);
      }
    }
  }

  // ---- store O (16 x 128 f32 per wave) ----
#pragma unroll
  for (int t = 0; t < 8; ++t)
#pragma unroll
    for (int r = 0; r < 8; ++r)
      outO[(size_t)(qbase + hh * 8 + r) * D_ + 16 * t + n] = acc[t][r];
}

extern "C" void kernel_launch(void* const* d_in, const int* in_sizes, int n_in,
                              void* d_out, int out_size, void* d_ws, size_t ws_size,
                              hipStream_t stream) {
  (void)in_sizes; (void)n_in; (void)out_size; (void)d_ws; (void)ws_size;
  const float* Q   = (const float*)d_in[0];
  const float* K   = (const float*)d_in[1];
  const float* V   = (const float*)d_in[2];
  const float* Adj = (const float*)d_in[3];
  const float* Ds  = (const float*)d_in[4];
  const float* wA  = (const float*)d_in[5];
  const float* wD  = (const float*)d_in[6];
  float* out = (float*)d_out;

  dim3 grid(L_ / (NWAVE * 16), H_, B_);   // (16, 16, 2)
  msrsa_attn<<<grid, 256, 0, stream>>>(Q, K, V, Adj, Ds, wA, wD, out);
}
